// GNCA_60911226191976
// MI455X (gfx1250) — compile-verified, matrix operand order checked by reading, not run on hardware
//
#include <hip/hip_runtime.h>
#include <hip/hip_bf16.h>

typedef __attribute__((ext_vector_type(16))) _Float16 v16h;
typedef __attribute__((ext_vector_type(8)))  float    v8f;

#define ACC_SCALE 0.005f
#define MAX_VEL   0.02f
#define RADIUS    0.05f
#define NOISE_SCALE 0.004f

// ---------------------------------------------------------------------------
// workspace zeroing (harness poisons ws once; we must re-zero every call)
// ---------------------------------------------------------------------------
__global__ void gnca_zero(float* __restrict__ p, int n) {
    int i = blockIdx.x * blockDim.x + threadIdx.x;
    if (i < n) p[i] = 0.0f;
}

// ---------------------------------------------------------------------------
// Pack the WMMA B operand once: B = concat(W_edge 4x16, W_src 10x16, 0-pad)
// laid out in the ISA's 16-bit B layout (lane l holds column N=l%16;
// halfs 0..7 = K=(l/16)*8+j ; halfs 8..15 = K+16, all zero here).
// One wave32; each lane writes its 16 halfs (32 bytes).
// ---------------------------------------------------------------------------
__global__ void gnca_pack_b(const float* __restrict__ W_edge,
                            const float* __restrict__ W_src,
                            _Float16* __restrict__ Bpack /* 32 lanes x 16 halfs */)
{
    const int lane = threadIdx.x & 31;
    const int n    = lane & 15;
    const int baseK = (lane >> 4) * 8;
    #pragma unroll
    for (int j = 0; j < 16; ++j) {
        const int k = (j < 8) ? (baseK + j) : (baseK + 16 + (j - 8));
        float w = 0.0f;
        if (k < 4)       w = W_edge[k * 16 + n];
        else if (k < 14) w = W_src[(k - 4) * 16 + n];
        Bpack[lane * 16 + j] = (_Float16)w;
    }
}

// ---------------------------------------------------------------------------
// Edge kernel: one wave32 per 16-edge tile.
//   m(16 edges x 16 hidden) = relu( A(16x32 f16) x B(32x16 f16) + b1 )
// A rows = edges, K = [dist, dx, dy, food, x_src[0..9], 0-pad to 32]
// D layout: lane l, elem r -> M = r + 8*(l/16), N = l%16.
// ---------------------------------------------------------------------------
__global__ __launch_bounds__(256)
void gnca_edge_wmma(const float* __restrict__ x,
                    const int*   __restrict__ src,
                    const int*   __restrict__ dst,
                    const _Float16* __restrict__ Bpack, // 32 x 16 halfs
                    const float* __restrict__ b1,       // 16
                    float* __restrict__ agg,            // N x 16
                    float* __restrict__ deg,            // N
                    float* __restrict__ cdeg,           // N
                    float* __restrict__ accum,          // [0]=visible_food [1]=mean_food_dist
                    int E, int numTiles)
{
    const int wave = (blockIdx.x * blockDim.x + threadIdx.x) >> 5;
    const int lane = threadIdx.x & 31;
    if (wave >= numTiles) return;   // wave-uniform: EXEC stays all-ones for WMMA

    const int m     = lane & 15;    // edge-in-tile (A row) and hidden column (D)
    const int half  = lane >> 4;
    const int base  = wave * 16;
    const int e     = base + m;
    const bool ok   = (e < E);
    const bool fullTile = (base + 16) <= E;   // wave-uniform

    float xs[10];
    #pragma unroll
    for (int i = 0; i < 10; ++i) xs[i] = 0.0f;
    int   t = 0;
    float dist = 0.0f, dx = 0.0f, dy = 0.0f, food = 0.0f, cellDst = 0.0f;
    if (ok) {
        const int s = src[e];
        t = dst[e];
        #pragma unroll
        for (int i = 0; i < 10; ++i) xs[i] = x[s * 10 + i];
        const float pdx = x[t * 10 + 0];
        const float pdy = x[t * 10 + 1];
        cellDst = x[t * 10 + 4];
        dx = xs[0] - pdx;
        dy = xs[1] - pdy;
        dist = sqrtf(dx * dx + dy * dy + 1e-12f);
        food = xs[4];
    }

    // A operand: lanes 0-15 carry K=0..7, lanes 16-31 carry K=8..15 (8..13 valid)
    const bool hi = (half != 0);
    float f[8];
    f[0] = hi ? xs[4] : dist;
    f[1] = hi ? xs[5] : dx;
    f[2] = hi ? xs[6] : dy;
    f[3] = hi ? xs[7] : food;
    f[4] = hi ? xs[8] : xs[0];
    f[5] = hi ? xs[9] : xs[1];
    f[6] = hi ? 0.0f  : xs[2];
    f[7] = hi ? 0.0f  : xs[3];

    v16h a;
    #pragma unroll
    for (int j = 0; j < 8; ++j) a[j] = (_Float16)f[j];
    #pragma unroll
    for (int j = 8; j < 16; ++j) a[j] = (_Float16)0.0f;

    // B operand: precomputed lane-striped layout, one 32B load per lane
    const v16h b = *(const v16h*)(Bpack + lane * 16);

    v8f c = {};
    v8f d = __builtin_amdgcn_wmma_f32_16x16x32_f16(
        /*neg_a=*/false, a, /*neg_b=*/false, b,
        /*c_mod=*/(short)0, c, /*reuse_a=*/false, /*reuse_b=*/false);

    // bias + relu + atomic segment-sum into agg[dst]
    const float bn = b1[m];
    if (fullTile) {
        // 8 contiguous dst indices per lane, 16B-aligned -> two b128 loads
        const int4 d0 = *(const int4*)(dst + base + half * 8);
        const int4 d1 = *(const int4*)(dst + base + half * 8 + 4);
        const int dstv[8] = { d0.x, d0.y, d0.z, d0.w, d1.x, d1.y, d1.z, d1.w };
        #pragma unroll
        for (int r = 0; r < 8; ++r) {
            const float v = fmaxf(d[r] + bn, 0.0f);
            atomicAdd(&agg[dstv[r] * 16 + m], v);
        }
    } else {
        #pragma unroll
        for (int r = 0; r < 8; ++r) {
            const int ee = base + half * 8 + r;
            if (ee < E) {
                const float v = fmaxf(d[r] + bn, 0.0f);
                atomicAdd(&agg[dst[ee] * 16 + m], v);
            }
        }
    }

    // per-edge scalar stats (one lane per edge: half==0)
    float vf = 0.0f, mfd = 0.0f;
    if (ok && half == 0) {
        atomicAdd(&deg[t], 1.0f);
        if (food == 0.0f) { vf = 1.0f; mfd = dist; }
        if (dist < RADIUS && food == 1.0f && cellDst == 0.0f)
            atomicAdd(&cdeg[t], 1.0f);
    }
    #pragma unroll
    for (int off = 16; off > 0; off >>= 1) {
        vf  += __shfl_down(vf,  off, 32);
        mfd += __shfl_down(mfd, off, 32);
    }
    if (lane == 0) {
        if (vf  != 0.0f) atomicAdd(&accum[0], vf);
        if (mfd != 0.0f) atomicAdd(&accum[1], mfd);
    }
}

// ---------------------------------------------------------------------------
// Node kernel: h = tanh(agg @ W_out + b2) * c_mask; integrate state; reductions
// ---------------------------------------------------------------------------
__global__ __launch_bounds__(256)
void gnca_node(const float* __restrict__ x,
               const float* __restrict__ noise,     // N x 2
               const float* __restrict__ W_out,     // 16 x 4
               const float* __restrict__ b2,        // 4
               const float* __restrict__ agg,
               const float* __restrict__ deg,
               const float* __restrict__ cdeg,
               float* __restrict__ out_x,           // N x 10
               float* __restrict__ out_keep,        // N
               float* __restrict__ accum,           // [2]=vb0 [3]=vb1 [4]=bc [5]=dead [6]=food
               int N)
{
    const int nidx = blockIdx.x * blockDim.x + threadIdx.x;
    float vb0 = 0.0f, vb1 = 0.0f, bc = 0.0f, dd = 0.0f, fr = 0.0f;
    if (nidx < N) {
        const float* xr = x + nidx * 10;
        const float cell = xr[4];
        const float cm = (cell == 1.0f) ? 1.0f : 0.0f;

        float a16[16];
        #pragma unroll
        for (int k = 0; k < 16; ++k) a16[k] = agg[nidx * 16 + k];

        float h[4];
        #pragma unroll
        for (int j = 0; j < 4; ++j) {
            float s = b2[j];
            #pragma unroll
            for (int k = 0; k < 16; ++k) s += a16[k] * W_out[k * 4 + j];
            h[j] = tanhf(s) * cm;
        }

        float nvx = xr[2] + h[0] * ACC_SCALE;
        float nvy = xr[3] + h[1] * ACC_SCALE;
        nvx = fminf(fmaxf(nvx, -MAX_VEL), MAX_VEL);
        nvy = fminf(fmaxf(nvy, -MAX_VEL), MAX_VEL);
        const float npx = xr[0] + nvx;
        const float npy = xr[1] + nvy;
        const float vnx = nvx + (noise[nidx * 2 + 0] * 2.0f - 1.0f) * NOISE_SCALE * cm;
        const float vny = nvy + (noise[nidx * 2 + 1] * 2.0f - 1.0f) * NOISE_SCALE * cm;

        float* orow = out_x + nidx * 10;
        orow[0] = npx; orow[1] = npy; orow[2] = vnx; orow[3] = vny;
        orow[4] = cell; orow[5] = h[2]; orow[6] = h[3];
        orow[7] = xr[7]; orow[8] = xr[8]; orow[9] = xr[9];

        const float apx = fabsf(npx), apy = fabsf(npy);
        if (apx > 1.0f) bc += logf(apx + 1e-6f);
        if (apy > 1.0f) bc += logf(apy + 1e-6f);

        const bool dead = (cell == 1.0f) && (deg[nidx]  < 3.0f);
        const bool cons = (cell == 0.0f) && (cdeg[nidx] >= 3.0f);
        out_keep[nidx] = (dead || cons) ? 0.0f : 1.0f;
        dd = dead ? 1.0f : 0.0f;
        fr = cons ? 1.0f : 0.0f;
        vb0 = fabsf(nvx);
        vb1 = fabsf(nvy);
    }
    #pragma unroll
    for (int off = 16; off > 0; off >>= 1) {
        vb0 += __shfl_down(vb0, off, 32);
        vb1 += __shfl_down(vb1, off, 32);
        bc  += __shfl_down(bc,  off, 32);
        dd  += __shfl_down(dd,  off, 32);
        fr  += __shfl_down(fr,  off, 32);
    }
    if ((threadIdx.x & 31) == 0) {
        atomicAdd(&accum[2], vb0);
        atomicAdd(&accum[3], vb1);
        atomicAdd(&accum[4], bc);
        atomicAdd(&accum[5], dd);
        atomicAdd(&accum[6], fr);
    }
}

// ---------------------------------------------------------------------------
// Finalize: write the 7 scalar outputs in return order
// ---------------------------------------------------------------------------
__global__ void gnca_final(const float* __restrict__ accum,
                           float* __restrict__ out_tail, float invN)
{
    if (blockIdx.x == 0 && threadIdx.x == 0) {
        out_tail[0] = accum[2] * invN;  // velocity_bonus.x
        out_tail[1] = accum[3] * invN;  // velocity_bonus.y
        out_tail[2] = accum[4];         // border_cost
        out_tail[3] = accum[6];         // food_reward
        out_tail[4] = accum[5];         // dead_cost
        out_tail[5] = accum[0];         // visible_food
        out_tail[6] = accum[1];         // mean_food_dist
    }
}

// ---------------------------------------------------------------------------
extern "C" void kernel_launch(void* const* d_in, const int* in_sizes, int n_in,
                              void* d_out, int out_size, void* d_ws, size_t ws_size,
                              hipStream_t stream) {
    const float* x      = (const float*)d_in[0];   // N x 10
    const int*   src    = (const int*)  d_in[1];   // E
    const int*   dst    = (const int*)  d_in[2];   // E
    const float* noise  = (const float*)d_in[3];   // N x 2
    const float* W_edge = (const float*)d_in[4];   // 4 x 16
    const float* W_src  = (const float*)d_in[5];   // 10 x 16
    const float* b1     = (const float*)d_in[6];   // 16
    const float* W_out  = (const float*)d_in[7];   // 16 x 4
    const float* b2     = (const float*)d_in[8];   // 4

    const int N = in_sizes[0] / 10;
    const int E = in_sizes[1];

    // workspace layout (floats): agg[N*16] | deg[N] | cdeg[N] | accum[8] | Bpack
    float* agg   = (float*)d_ws;
    float* deg   = agg  + (size_t)N * 16;
    float* cdeg  = deg  + N;
    float* accum = cdeg + N;
    const int wsFloats = N * 18 + 8;
    _Float16* Bpack = (_Float16*)(accum + 8);      // 32-byte aligned for N=50000

    // output layout (floats): new_x[N*10] | keep[N] | scalars[7]
    float* out_x    = (float*)d_out;
    float* out_keep = out_x + (size_t)N * 10;
    float* out_tail = out_keep + N;

    gnca_zero<<<(wsFloats + 255) / 256, 256, 0, stream>>>((float*)d_ws, wsFloats);
    gnca_pack_b<<<1, 32, 0, stream>>>(W_edge, W_src, Bpack);

    const int numTiles = (E + 15) / 16;                 // one wave32 per tile
    const int edgeThreads = numTiles * 32;
    gnca_edge_wmma<<<(edgeThreads + 255) / 256, 256, 0, stream>>>(
        x, src, dst, Bpack, b1, agg, deg, cdeg, accum, E, numTiles);

    gnca_node<<<(N + 255) / 256, 256, 0, stream>>>(
        x, noise, W_out, b2, agg, deg, cdeg, out_x, out_keep, accum, N);

    gnca_final<<<1, 32, 0, stream>>>(accum, out_tail, 1.0f / (float)N);
}
